// Decoder_78125455114561
// MI455X (gfx1250) — matrix operand order repaired and checked by reading, not compiled, not run on hardware
//
#include <hip/hip_runtime.h>
#include <hip/hip_bf16.h>

// MI455X (gfx1250) LSTM decoder rollout.
// - bf16 WMMA GEMMs (v_wmma_f32_16x16x32_bf16), f32 accumulate
// - W_out bf16-resident in 192MB L2 across the 20 sequential vocab GEMMs
// - LSTM gates fused into the cell-GEMM epilogue (registers only; h ping-pong)
// - A-operand LDS staging via global_load_async_to_lds_b128 + s_wait_asynccnt

typedef __attribute__((ext_vector_type(16))) __bf16 v16bf;
typedef __attribute__((ext_vector_type(8)))  float  v8f;

#define B_DIM 64
#define H_DIM 512
#define G_DIM 2048   /* 4*H */
#define V_DIM 32000

__device__ __forceinline__ unsigned short f32_bf16(float f) {
  unsigned int u = __float_as_uint(f);
  u += 0x7FFFu + ((u >> 16) & 1u);      // round-to-nearest-even
  return (unsigned short)(u >> 16);
}

// generic -> LDS(AS3) offset for async-to-LDS destinations
__device__ __forceinline__ unsigned lds_off(const void* p) {
  return (unsigned)(size_t)(__attribute__((address_space(3))) const char*)p;
}

// async copy of 16B global -> LDS (GV mode), tracked by ASYNCcnt
__device__ __forceinline__ void async_b128(unsigned ldsdst, const void* gsrc) {
  asm volatile("global_load_async_to_lds_b128 %0, %1, off"
               :: "v"(ldsdst), "v"(gsrc) : "memory");
}
__device__ __forceinline__ void wait_async0() {
  asm volatile("s_wait_asynccnt 0x0" ::: "memory");
}

// ---------------------------------------------------------------------------
// Fused LSTM cell:
//   g = A0[64,512] @ Wih[2048,512]^T + Hprev[64,512] @ Whh[2048,512]^T + bsum
//   c' = sig(f)*c + sig(i)*tanh(g);  h' = sig(o)*tanh(c')  (computed in regs)
// A0: gather emb[words] (f32 -> bf16, row 0 zero) if emb != nullptr, else A0bf.
// Block = 256 thr = 8 waves, block tile = 64(M) x 32(j); each wave holds all
// four gate tiles for one (mt, jt) block. Grid = H/32 = 16. hout != Hprev.
// ---------------------------------------------------------------------------
__global__ __launch_bounds__(256)
void lstm_cell_wmma(const float* __restrict__ emb,
                    const int* __restrict__ words,
                    const unsigned short* __restrict__ A0bf,
                    const unsigned short* __restrict__ Wih,
                    const unsigned short* __restrict__ Hprev,
                    const unsigned short* __restrict__ Whh,
                    const float* __restrict__ bsum,
                    float* __restrict__ c,
                    unsigned short* __restrict__ hout)
{
  __shared__ unsigned short sA[64 * 136];   // 64 rows x 128-half chunk, +8 pad

  const int tid  = threadIdx.x;
  const int lane = tid & 31;
  const int wv   = tid >> 5;
  const int mt   = wv >> 1;        // M tile 0..3
  const int jt   = wv & 1;         // j sub-tile within block's 32 cols
  const int ln   = lane & 15;
  const int hi   = lane >> 4;
  const int kb   = hi << 3;        // unused; A and B both use ko
  const int ko   = hi << 4;        // B-fragment K offset
  const int j0   = blockIdx.x * 32;
  const int jcol = j0 + jt * 16 + ln;       // 0..511

  v8f acc[4];                      // i, f, g, o accumulators (same (m,j) map)
#pragma unroll
  for (int g = 0; g < 4; ++g) {
    float bv = bsum[g * H_DIM + jcol];
#pragma unroll
    for (int r = 0; r < 8; ++r) acc[g][r] = bv;
  }

  const int srow = tid >> 2;        // staging row 0..63
  const int sq   = (tid & 3) * 32;  // halves [sq, sq+32) within the chunk

  for (int p = 0; p < 2; ++p) {
    const unsigned short* Abf = (p == 0) ? A0bf : Hprev;
    const unsigned short* Bw  = (p == 0) ? Wih  : Whh;
    const bool gather = (p == 0) && (emb != nullptr);
    const unsigned short* brow[4];
#pragma unroll
    for (int g = 0; g < 4; ++g)
      brow[g] = Bw + (size_t)(g * H_DIM + jcol) * H_DIM;

    for (int cb = 0; cb < H_DIM; cb += 128) {
      __syncthreads();
      if (gather) {
        int word = words[srow];
        const float4* src = (const float4*)(emb + (size_t)word * H_DIM + cb + sq);
#pragma unroll
        for (int u = 0; u < 8; ++u) {
          float4 v = make_float4(0.f, 0.f, 0.f, 0.f);
          if (word != 0) v = src[u];           // padding_idx=0 -> zero row
          unsigned short* d = &sA[srow * 136 + sq + u * 4];
          d[0] = f32_bf16(v.x); d[1] = f32_bf16(v.y);
          d[2] = f32_bf16(v.z); d[3] = f32_bf16(v.w);
        }
      } else {
        const unsigned short* gsrc = Abf + (size_t)srow * H_DIM + cb + sq;
        unsigned ldst = lds_off(&sA[srow * 136 + sq]);
#pragma unroll
        for (int u = 0; u < 4; ++u) async_b128(ldst + u * 16, gsrc + u * 8);
        wait_async0();
      }
      __syncthreads();

      const __bf16* ap = (const __bf16*)&sA[(mt * 16 + ln) * 136];
#pragma unroll
      for (int kl = 0; kl < 128; kl += 32) {
        v16bf a;                    // A 16x32: e -> K=e+16*hi
#pragma unroll
        for (int e = 0; e < 16; ++e)
          a[e] = ap[kl + ko + e];
#pragma unroll
        for (int g = 0; g < 4; ++g) {
          v16bf bf = *(const v16bf*)(brow[g] + cb + kl + ko);
          acc[g] = __builtin_amdgcn_wmma_f32_16x16x32_bf16(
              false, a, false, bf, (short)0, acc[g], false, false);
        }
      }
    }
  }

  // gate epilogue, all in registers: C/D layout -> M = mt*16+8*hi+r, N = jcol
#pragma unroll
  for (int r = 0; r < 8; ++r) {
    int m   = mt * 16 + hi * 8 + r;
    int idx = m * H_DIM + jcol;
    float si = 1.f / (1.f + __expf(-acc[0][r]));
    float sf = 1.f / (1.f + __expf(-acc[1][r]));
    float so = 1.f / (1.f + __expf(-acc[3][r]));
    float c2 = sf * c[idx] + si * tanhf(acc[2][r]);
    float h2 = so * tanhf(c2);
    c[idx]    = c2;
    hout[idx] = f32_bf16(h2);
  }
}

// ---------------------------------------------------------------------------
// Logits GEMM: C[64,N] = A[64,512] @ B[N,512]^T + bias[N]   (bf16 in, f32 out)
// Block = 256 thr = 8 waves, tile 64x64; grid.x = N/64 = 500.
// ---------------------------------------------------------------------------
__global__ __launch_bounds__(256)
void gemm64_wmma(const unsigned short* __restrict__ Abf,
                 const unsigned short* __restrict__ Bw,
                 const float* __restrict__ bias,
                 float* __restrict__ C, long long crstride)
{
  __shared__ unsigned short sA[64 * 136];

  const int tid  = threadIdx.x;
  const int lane = tid & 31;
  const int wv   = tid >> 5;
  const int mt   = wv >> 1;
  const int nt0  = (wv & 1) * 2;
  const int ln   = lane & 15;
  const int hi   = lane >> 4;
  const int kb   = hi << 3;
  const int ko   = hi << 4;
  const long long nbase = (long long)blockIdx.x * 64;

  v8f acc0, acc1;
  {
    float bv0 = bias[nbase + nt0 * 16 + ln];
    float bv1 = bias[nbase + (nt0 + 1) * 16 + ln];
#pragma unroll
    for (int r = 0; r < 8; ++r) { acc0[r] = bv0; acc1[r] = bv1; }
  }

  const int srow = tid >> 2;
  const int sq   = (tid & 3) * 32;
  const unsigned short* brow0 = Bw + (size_t)(nbase + nt0 * 16 + ln) * H_DIM;
  const unsigned short* brow1 = Bw + (size_t)(nbase + (nt0 + 1) * 16 + ln) * H_DIM;

  for (int cb = 0; cb < H_DIM; cb += 128) {
    __syncthreads();
    {
      const unsigned short* gsrc = Abf + (size_t)srow * H_DIM + cb + sq;
      unsigned ldst = lds_off(&sA[srow * 136 + sq]);
#pragma unroll
      for (int u = 0; u < 4; ++u) async_b128(ldst + u * 16, gsrc + u * 8);
      wait_async0();
    }
    __syncthreads();

    const __bf16* ap = (const __bf16*)&sA[(mt * 16 + ln) * 136];
#pragma unroll
    for (int kl = 0; kl < 128; kl += 32) {
      v16bf a;
#pragma unroll
      for (int e = 0; e < 16; ++e)
        a[e] = ap[kl + ko + e];
      v16bf b0f = *(const v16bf*)(brow0 + cb + kl + ko);
      v16bf b1f = *(const v16bf*)(brow1 + cb + kl + ko);
      acc0 = __builtin_amdgcn_wmma_f32_16x16x32_bf16(
          false, a, false, b0f, (short)0, acc0, false, false);
      acc1 = __builtin_amdgcn_wmma_f32_16x16x32_bf16(
          false, a, false, b1f, (short)0, acc1, false, false);
    }
  }

#pragma unroll
  for (int r = 0; r < 8; ++r) {
    long long m = (long long)mt * 16 + hi * 8 + r;
    C[m * crstride + nbase + nt0 * 16 + ln]       = acc0[r];
    C[m * crstride + nbase + (nt0 + 1) * 16 + ln] = acc1[r];
  }
}

// ---------------------------------------------------------------------------
// Per-row argmax over V (first-occurrence ties). Grid: B x 256.
// ---------------------------------------------------------------------------
__global__ __launch_bounds__(256)
void argmax_k(const float* __restrict__ logits, long long rstride,
              int* __restrict__ words)
{
  __shared__ float sv[256];
  __shared__ int   sidx[256];
  const int tid = threadIdx.x;
  const float* row = logits + (long long)blockIdx.x * rstride;
  float bv = -3.402823466e38f;
  int bi = 0x7fffffff;
  for (int v = tid; v < V_DIM; v += 256) {
    float x = row[v];
    if (x > bv) { bv = x; bi = v; }
  }
  sv[tid] = bv; sidx[tid] = bi;
  __syncthreads();
  for (int s = 128; s > 0; s >>= 1) {
    if (tid < s) {
      if (sv[tid + s] > sv[tid] ||
          (sv[tid + s] == sv[tid] && sidx[tid + s] < sidx[tid])) {
        sv[tid] = sv[tid + s]; sidx[tid] = sidx[tid + s];
      }
    }
    __syncthreads();
  }
  if (tid == 0) words[blockIdx.x] = sidx[0];
}

// ---------------------------------------------------------------------------
// In-place log_softmax over each V-row of out[B*T, V]. Grid: B*T x 256.
// ---------------------------------------------------------------------------
__global__ __launch_bounds__(256)
void logsoftmax_k(float* __restrict__ out)
{
  __shared__ float red[256];
  const int tid = threadIdx.x;
  float* row = out + (size_t)blockIdx.x * V_DIM;

  float m = -3.402823466e38f;
  for (int v = tid; v < V_DIM; v += 256) m = fmaxf(m, row[v]);
  red[tid] = m; __syncthreads();
  for (int s = 128; s > 0; s >>= 1) {
    if (tid < s) red[tid] = fmaxf(red[tid], red[tid + s]);
    __syncthreads();
  }
  m = red[0]; __syncthreads();

  float acc = 0.f;
  for (int v = tid; v < V_DIM; v += 256) acc += __expf(row[v] - m);
  red[tid] = acc; __syncthreads();
  for (int s = 128; s > 0; s >>= 1) {
    if (tid < s) red[tid] += red[tid + s];
    __syncthreads();
  }
  float lse = m + __logf(red[0]);

  for (int v = tid; v < V_DIM; v += 256) row[v] = row[v] - lse;
}

// ------------------------- small prep kernels ------------------------------
__global__ void cvt_bf16_k(const float* __restrict__ src,
                           unsigned short* __restrict__ dst, int n)
{
  int i = blockIdx.x * 256 + threadIdx.x;
  if (i < n) dst[i] = f32_bf16(src[i]);
}

__global__ void bias_sum_k(const float* __restrict__ a,
                           const float* __restrict__ b,
                           float* __restrict__ o, int n)
{
  int i = blockIdx.x * 256 + threadIdx.x;
  if (i < n) o[i] = a[i] + b[i];
}

__global__ void init_words_k(int* __restrict__ words, int start)
{
  if (threadIdx.x < B_DIM) words[threadIdx.x] = start;
}

// ---------------------------------------------------------------------------
extern "C" void kernel_launch(void* const* d_in, const int* in_sizes, int n_in,
                              void* d_out, int out_size, void* d_ws, size_t ws_size,
                              hipStream_t stream)
{
  (void)in_sizes; (void)n_in; (void)ws_size;
  const float* h_in = (const float*)d_in[1];
  const float* c_in = (const float*)d_in[2];
  const float* emb  = (const float*)d_in[3];
  const float* Wih0 = (const float*)d_in[4];
  const float* Whh0 = (const float*)d_in[5];
  const float* bih0 = (const float*)d_in[6];
  const float* bhh0 = (const float*)d_in[7];
  const float* Wih1 = (const float*)d_in[8];
  const float* Whh1 = (const float*)d_in[9];
  const float* bih1 = (const float*)d_in[10];
  const float* bhh1 = (const float*)d_in[11];
  const float* Wout = (const float*)d_in[12];
  const float* bout = (const float*)d_in[13];
  float* out = (float*)d_out;
  const int T = out_size / (B_DIM * V_DIM);      // 20 in reference setup

  // ---- workspace carve-out (256B aligned; total ~42 MB) ----
  char* ws = (char*)d_ws;
  size_t off = 0;
  auto carve = [&](size_t bytes) -> void* {
    void* p = ws + off;
    off = (off + bytes + 255) & ~(size_t)255;
    return p;
  };
  const int BH = B_DIM * H_DIM;
  unsigned short* wout_bf = (unsigned short*)carve((size_t)V_DIM * H_DIM * 2);
  unsigned short* wih0_bf = (unsigned short*)carve((size_t)G_DIM * H_DIM * 2);
  unsigned short* whh0_bf = (unsigned short*)carve((size_t)G_DIM * H_DIM * 2);
  unsigned short* wih1_bf = (unsigned short*)carve((size_t)G_DIM * H_DIM * 2);
  unsigned short* whh1_bf = (unsigned short*)carve((size_t)G_DIM * H_DIM * 2);
  float* bsum0 = (float*)carve((size_t)G_DIM * 4);
  float* bsum1 = (float*)carve((size_t)G_DIM * 4);
  unsigned short* hbf = (unsigned short*)carve((size_t)4 * BH * 2); // [l][par]
  float* cws  = (float*)carve((size_t)2 * BH * 4);
  int* wordsb = (int*)carve(256);

  // ---- preprocessing (deterministic, re-done every call) ----
  const int NW = V_DIM * H_DIM;
  const int NG = G_DIM * H_DIM;
  cvt_bf16_k<<<(NW + 255) / 256, 256, 0, stream>>>(Wout, wout_bf, NW);
  cvt_bf16_k<<<(NG + 255) / 256, 256, 0, stream>>>(Wih0, wih0_bf, NG);
  cvt_bf16_k<<<(NG + 255) / 256, 256, 0, stream>>>(Whh0, whh0_bf, NG);
  cvt_bf16_k<<<(NG + 255) / 256, 256, 0, stream>>>(Wih1, wih1_bf, NG);
  cvt_bf16_k<<<(NG + 255) / 256, 256, 0, stream>>>(Whh1, whh1_bf, NG);
  bias_sum_k<<<G_DIM / 256, 256, 0, stream>>>(bih0, bhh0, bsum0, G_DIM);
  bias_sum_k<<<G_DIM / 256, 256, 0, stream>>>(bih1, bhh1, bsum1, G_DIM);
  cvt_bf16_k<<<(BH + 255) / 256, 256, 0, stream>>>(h_in, hbf, BH);            // l0 par0
  cvt_bf16_k<<<(BH + 255) / 256, 256, 0, stream>>>(h_in + BH, hbf + 2 * BH, BH); // l1 par0
  hipMemcpyAsync(cws, c_in, (size_t)2 * BH * 4, hipMemcpyDeviceToDevice, stream);
  init_words_k<<<1, 64, 0, stream>>>(wordsb, 1 /* START */);

  // ---- sequential rollout: 4 kernels per step ----
  for (int t = 0; t < T; ++t) {
    const int rd = t & 1, wr = rd ^ 1;          // h ping-pong (WAR-safe)
    unsigned short* h0r = hbf + rd * BH;
    unsigned short* h0w = hbf + wr * BH;
    unsigned short* h1r = hbf + 2 * BH + rd * BH;
    unsigned short* h1w = hbf + 2 * BH + wr * BH;

    lstm_cell_wmma<<<H_DIM / 32, 256, 0, stream>>>(
        emb, wordsb, nullptr, wih0_bf, h0r, whh0_bf, bsum0, cws, h0w);
    lstm_cell_wmma<<<H_DIM / 32, 256, 0, stream>>>(
        nullptr, nullptr, h0w, wih1_bf, h1r, whh1_bf, bsum1, cws + BH, h1w);
    gemm64_wmma<<<V_DIM / 64, 256, 0, stream>>>(
        h1w, wout_bf, bout, out + (size_t)t * V_DIM, (long long)T * V_DIM);
    argmax_k<<<B_DIM, 256, 0, stream>>>(out + (size_t)t * V_DIM,
                                        (long long)T * V_DIM, wordsb);
  }

  // in-place log_softmax over all B*T rows
  logsoftmax_k<<<B_DIM * T, 256, 0, stream>>>(out);
}
